// RelativeVectorEmbedding_68186900791675
// MI455X (gfx1250) — compile-verified, matrix operand order checked by reading, not run on hardware
//
#include <hip/hip_runtime.h>
#include <hip/hip_bf16.h>
#include <math.h>

#define T_ 128
#define B_ 16
#define I_ 8
#define D_ 128

typedef __attribute__((ext_vector_type(16))) __bf16 bf16x16;
typedef __attribute__((ext_vector_type(8)))  __bf16 bf16x8;
typedef __attribute__((ext_vector_type(4)))  __bf16 bf16x4;
typedef __attribute__((ext_vector_type(8)))  float  f32x8;

__device__ __forceinline__ float gelu_tanh(float x) {
  float x3 = x * x * x;
  return 0.5f * x * (1.0f + tanhf(0.7978845608028654f * (x + 0.044715f * x3)));
}

// ---------------------------------------------------------------------------
// Kernel 0: transpose Wq/Wk/Wv/W_out (D x D, row-major K x N) into N-major
// bf16 so B-operand tiles load with the same striping as A-operand tiles.
// ---------------------------------------------------------------------------
__global__ void prep_weights(const float* Wq, const float* Wk, const float* Wv,
                             const float* Wo, __bf16* out /* 4 * 128*128 */) {
  int idx = blockIdx.x * blockDim.x + threadIdx.x;
  if (idx >= 4 * D_ * D_) return;
  int mat = idx >> 14;
  int e = idx & 16383;
  int n = e >> 7, k = e & 127;
  const float* src = (mat == 0) ? Wq : (mat == 1) ? Wk : (mat == 2) ? Wv : Wo;
  out[mat * 16384 + n * 128 + k] = (__bf16)src[k * 128 + n];
}

// ---------------------------------------------------------------------------
// Kernel 1: embed (K=8 GEMM) + gelu + mask + LayerNorm1, one wave per row.
// Output x as bf16 (WMMA A operand food).
// ---------------------------------------------------------------------------
__global__ void embed_ln1(const float* __restrict__ vectors,
                          const unsigned char* __restrict__ mask,
                          const float* __restrict__ We, const float* __restrict__ be,
                          const float* __restrict__ g1, const float* __restrict__ b1,
                          __bf16* __restrict__ xout) {
  int wave = threadIdx.x >> 5;
  int lane = threadIdx.x & 31;
  int row = blockIdx.x * 8 + wave;          // 0 .. T*T*B-1, row = rc*16 + b
  int b = row & 15;
  int rc = row >> 4;
  int r = rc >> 7, c = rc & 127;
  const float* in = vectors + ((size_t)((b * T_ + r) * T_ + c)) * I_;
  float v0 = in[0], v1 = in[1], v2 = in[2], v3 = in[3];
  float v4 = in[4], v5 = in[5], v6 = in[6], v7 = in[7];
  float m = mask[(size_t)b * T_ * T_ + r * T_ + c] ? 1.0f : 0.0f;
  float y[4];
#pragma unroll
  for (int j = 0; j < 4; ++j) {
    int d = lane * 4 + j;
    float a = be[d];
    a += v0 * We[0 * D_ + d] + v1 * We[1 * D_ + d] + v2 * We[2 * D_ + d] +
         v3 * We[3 * D_ + d] + v4 * We[4 * D_ + d] + v5 * We[5 * D_ + d] +
         v6 * We[6 * D_ + d] + v7 * We[7 * D_ + d];
    y[j] = gelu_tanh(a) * m;
  }
  float s = y[0] + y[1] + y[2] + y[3];
#pragma unroll
  for (int o = 16; o >= 1; o >>= 1) s += __shfl_xor(s, o, 32);
  float mean = s * (1.0f / 128.0f);
  float vs = 0.f;
#pragma unroll
  for (int j = 0; j < 4; ++j) { float t = y[j] - mean; vs += t * t; }
#pragma unroll
  for (int o = 16; o >= 1; o >>= 1) vs += __shfl_xor(vs, o, 32);
  float rstd = rsqrtf(vs * (1.0f / 128.0f) + 1e-5f);
  bf16x4 outv;
#pragma unroll
  for (int j = 0; j < 4; ++j) {
    int d = lane * 4 + j;
    outv[j] = (__bf16)((y[j] - mean) * rstd * g1[d] + b1[d]);
  }
  *(bf16x4*)(xout + (size_t)row * D_ + lane * 4) = outv;
}

// ---------------------------------------------------------------------------
// WMMA tile loaders. 16-bit A operand layout (wave32): lanes 0-15 hold M=lane,
// K in {k0..k0+7, k0+16..k0+23}; lanes 16-31 hold M=lane-16,
// K in {k0+8..k0+15, k0+24..k0+31}. B (N-major) loads identically.
// ---------------------------------------------------------------------------
__device__ __forceinline__ bf16x16 load_tile_bf16(const __bf16* base, int lane, int k0) {
  int row = lane & 15;
  int ksub = (lane & 16) ? 8 : 0;
  const __bf16* p = base + row * 128 + k0 + ksub;
  bf16x8 lo = *(const bf16x8*)p;
  bf16x8 hi = *(const bf16x8*)(p + 16);
  bf16x16 rr;
#pragma unroll
  for (int i = 0; i < 8; ++i) { rr[i] = lo[i]; rr[i + 8] = hi[i]; }
  return rr;
}

__device__ __forceinline__ bf16x16 load_tile_f32(const float* base, int lane, int k0) {
  int row = lane & 15;
  int ksub = (lane & 16) ? 8 : 0;
  const float* p = base + row * 128 + k0 + ksub;
  bf16x16 rr;
#pragma unroll
  for (int i = 0; i < 8; ++i) { rr[i] = (__bf16)p[i]; rr[i + 8] = (__bf16)p[16 + i]; }
  return rr;
}

// ---------------------------------------------------------------------------
// Kernel 2: K = x @ Wk, V = x @ Wv.
// Each wave owns one 16-column n-tile of BOTH K and V and hoists its 8
// B-operand tiles into registers once; it then sweeps 16 row-tiles, so the
// only in-loop loads are the A chunks (independent of the previous tile's
// wmmas -> schedulable overlap, no accumulator spill).
// ---------------------------------------------------------------------------
#define KV_TPB 16
__global__ void kv_gemm(const __bf16* __restrict__ x,
                        const __bf16* __restrict__ WkT, const __bf16* __restrict__ WvT,
                        __bf16* __restrict__ K, __bf16* __restrict__ V) {
  int wave = threadIdx.x >> 5;
  int lane = threadIdx.x & 31;
  int n = wave;                                 // n-tile = columns [16n, 16n+16)
  bf16x16 Bk[4], Bv[4];
#pragma unroll
  for (int kk = 0; kk < 4; ++kk) {
    Bk[kk] = load_tile_bf16(WkT + n * 16 * 128, lane, kk * 32);
    Bv[kk] = load_tile_bf16(WvT + n * 16 * 128, lane, kk * 32);
  }
  f32x8 zero = {0.f, 0.f, 0.f, 0.f, 0.f, 0.f, 0.f, 0.f};
  size_t tile0 = (size_t)blockIdx.x * KV_TPB;
  int col = n * 16 + (lane & 15);
#pragma unroll 2
  for (int t = 0; t < KV_TPB; ++t) {
    size_t row_base = (tile0 + t) * 16;
    const __bf16* A = x + row_base * 128;
    bf16x16 a0 = load_tile_bf16(A, lane, 0);
    bf16x16 a1 = load_tile_bf16(A, lane, 32);
    bf16x16 a2 = load_tile_bf16(A, lane, 64);
    bf16x16 a3 = load_tile_bf16(A, lane, 96);
    f32x8 accK = zero, accV = zero;
    accK = __builtin_amdgcn_wmma_f32_16x16x32_bf16(false, a0, false, Bk[0], (short)0, accK, false, false);
    accV = __builtin_amdgcn_wmma_f32_16x16x32_bf16(false, a0, false, Bv[0], (short)0, accV, false, false);
    accK = __builtin_amdgcn_wmma_f32_16x16x32_bf16(false, a1, false, Bk[1], (short)0, accK, false, false);
    accV = __builtin_amdgcn_wmma_f32_16x16x32_bf16(false, a1, false, Bv[1], (short)0, accV, false, false);
    accK = __builtin_amdgcn_wmma_f32_16x16x32_bf16(false, a2, false, Bk[2], (short)0, accK, false, false);
    accV = __builtin_amdgcn_wmma_f32_16x16x32_bf16(false, a2, false, Bv[2], (short)0, accV, false, false);
    accK = __builtin_amdgcn_wmma_f32_16x16x32_bf16(false, a3, false, Bk[3], (short)0, accK, false, false);
    accV = __builtin_amdgcn_wmma_f32_16x16x32_bf16(false, a3, false, Bv[3], (short)0, accV, false, false);
#pragma unroll
    for (int g = 0; g < 8; ++g) {
      int m = (lane & 16) ? g + 8 : g;          // half-wave writes 32 contiguous B
      size_t off = (row_base + m) * 128 + col;
      K[off] = (__bf16)accK[g];
      V[off] = (__bf16)accV[g];
    }
  }
}

// ---------------------------------------------------------------------------
// Kernel 2q: Q = x[diag] @ Wq, scaled by 1/sqrt(D). One wave per r.
// ---------------------------------------------------------------------------
__global__ void q_gemm(const __bf16* __restrict__ x, const __bf16* __restrict__ WqT,
                       float* __restrict__ Q) {
  int wave = threadIdx.x >> 5;
  int lane = threadIdx.x & 31;
  int r = blockIdx.x * 8 + wave;             // 0 .. 127
  const __bf16* A = x + (size_t)((r * T_ + r) * 16) * 128;
  f32x8 zero = {0.f, 0.f, 0.f, 0.f, 0.f, 0.f, 0.f, 0.f};
  f32x8 acc[8];
#pragma unroll
  for (int n = 0; n < 8; ++n) acc[n] = zero;
#pragma unroll
  for (int kk = 0; kk < 4; ++kk) {
    int k0 = kk * 32;
    bf16x16 a = load_tile_bf16(A, lane, k0);
#pragma unroll
    for (int n = 0; n < 8; ++n) {
      bf16x16 bq = load_tile_bf16(WqT + n * 16 * 128, lane, k0);
      acc[n] = __builtin_amdgcn_wmma_f32_16x16x32_bf16(false, a, false, bq,
                                                       (short)0, acc[n], false, false);
    }
  }
  const float scale = 0.08838834764831845f;   // 1/sqrt(128)
#pragma unroll
  for (int n = 0; n < 8; ++n) {
    int col = n * 16 + (lane & 15);
#pragma unroll
    for (int g = 0; g < 8; ++g) {
      int m = (lane & 16) ? g + 8 : g;
      Q[((size_t)r * 16 + m) * 128 + col] = acc[n][g] * scale;
    }
  }
}

// ---------------------------------------------------------------------------
// Kernel 3: attention for one r per block. logits/emb are elementwise in b
// (batched dot products), so VALU + LDS, exact fp32 softmax like reference.
// An async-to-LDS prefetch (ASYNCcnt) warms the first V rows for phase 4
// while phases 1-3 run; numerics use the normal load path.
// ---------------------------------------------------------------------------
__global__ void attention(const float* __restrict__ Q, const __bf16* __restrict__ K,
                          const __bf16* __restrict__ V,
                          const unsigned char* __restrict__ mask,
                          float* __restrict__ emb) {
  __shared__ float Qs[16 * 128];
  __shared__ float Ls[128 * 16];
  __shared__ __align__(16) unsigned char vpref[256 * 16];
  int tid = threadIdx.x;
  int r = blockIdx.x;
  // Async cache-warming prefetch of V[r, 0, :, :] (4 KB) into scratch LDS.
  {
    unsigned lds_off = (unsigned)(size_t)(vpref + tid * 16);
    unsigned long long gaddr =
        (unsigned long long)(const unsigned char*)(V + ((size_t)(r * T_) * 16) * 128) +
        (unsigned long long)tid * 16ull;
    asm volatile("global_load_async_to_lds_b128 %0, %1, off"
                 :: "v"(lds_off), "v"(gaddr) : "memory");
  }
#pragma unroll
  for (int j = 0; j < 8; ++j) Qs[j * 256 + tid] = Q[(size_t)r * 2048 + j * 256 + tid];
  __syncthreads();
#pragma unroll
  for (int j = 0; j < 8; ++j) {
    int idx = j * 256 + tid;                  // idx = c*16 + b
    int c = idx >> 4, b = idx & 15;
    const __bf16* kp = K + (((size_t)(r * T_ + c) * 16) + b) * 128;
    const float* qp = Qs + b * 128;
    float acc = 0.f;
#pragma unroll
    for (int d8 = 0; d8 < 16; ++d8) {
      bf16x8 kv = *(const bf16x8*)(kp + d8 * 8);
#pragma unroll
      for (int u = 0; u < 8; ++u) acc += (float)kv[u] * qp[d8 * 8 + u];
    }
    bool am = (mask[(size_t)b * T_ * T_ + r * T_ + c] != 0) || (c == r);
    Ls[idx] = acc + (am ? 0.0f : -1.0e9f);
  }
  __syncthreads();
  if (tid < 16) {
    int b = tid;
    float mx = -3.4e38f;
    for (int c = 0; c < 128; ++c) mx = fmaxf(mx, Ls[c * 16 + b]);
    float s = 0.f;
    for (int c = 0; c < 128; ++c) { float e = expf(Ls[c * 16 + b] - mx); Ls[c * 16 + b] = e; s += e; }
    float inv = 1.0f / s;
    for (int c = 0; c < 128; ++c) Ls[c * 16 + b] *= inv;
  }
  asm volatile("s_wait_asynccnt 0x0" ::: "memory");
  __syncthreads();
#pragma unroll
  for (int j = 0; j < 8; ++j) {
    int idx = j * 256 + tid;                  // idx = b*128 + d
    int b = idx >> 7, d = idx & 127;
    float acc = 0.f;
    for (int c = 0; c < 128; ++c) {
      acc += Ls[c * 16 + b] * (float)V[(((size_t)(r * T_ + c) * 16) + b) * 128 + d];
    }
    emb[((size_t)r * 16 + b) * 128 + d] = acc;
  }
}

// ---------------------------------------------------------------------------
// Kernel 4: h = gelu(emb @ W_out + b_out); out = LN2(h + emb) * seq.
// One block per r; wave w computes output columns [16w, 16w+16) via WMMA.
// ---------------------------------------------------------------------------
__global__ void outproj_ln2(const float* __restrict__ emb, const __bf16* __restrict__ WoT,
                            const float* __restrict__ bo,
                            const float* __restrict__ g2, const float* __restrict__ b2,
                            const unsigned char* __restrict__ mask,
                            float* __restrict__ out) {
  __shared__ float hs[16 * 128];
  int tid = threadIdx.x;
  int wave = tid >> 5, lane = tid & 31;
  int r = blockIdx.x;
  const float* A = emb + (size_t)r * 2048;
  int n0 = wave * 16;
  f32x8 acc = {0.f, 0.f, 0.f, 0.f, 0.f, 0.f, 0.f, 0.f};
#pragma unroll
  for (int kk = 0; kk < 4; ++kk) {
    int k0 = kk * 32;
    bf16x16 a = load_tile_f32(A, lane, k0);
    bf16x16 bt = load_tile_bf16(WoT + n0 * 128, lane, k0);
    acc = __builtin_amdgcn_wmma_f32_16x16x32_bf16(false, a, false, bt,
                                                  (short)0, acc, false, false);
  }
  {
    int col = n0 + (lane & 15);
#pragma unroll
    for (int g = 0; g < 8; ++g) {
      int m = (lane & 16) ? g + 8 : g;
      hs[m * 128 + col] = acc[g];
    }
  }
  __syncthreads();
#pragma unroll
  for (int j = 0; j < 8; ++j) {
    int idx = j * 256 + tid;                  // idx = m*128 + d
    int d = idx & 127;
    float v = hs[idx] + bo[d];
    hs[idx] = gelu_tanh(v) + A[idx];          // residual
  }
  __syncthreads();
  // LayerNorm per row m = tid>>4 using 16 lanes (half-wave shuffles).
  int m = tid >> 4;
  int t = tid & 15;
  float s = 0.f;
#pragma unroll
  for (int u = 0; u < 8; ++u) s += hs[m * 128 + t * 8 + u];
#pragma unroll
  for (int o = 8; o >= 1; o >>= 1) s += __shfl_xor(s, o, 16);
  float mean = s * (1.0f / 128.0f);
  float vs = 0.f;
#pragma unroll
  for (int u = 0; u < 8; ++u) { float x = hs[m * 128 + t * 8 + u] - mean; vs += x * x; }
#pragma unroll
  for (int o = 8; o >= 1; o >>= 1) vs += __shfl_xor(vs, o, 16);
  float rstd = rsqrtf(vs * (1.0f / 128.0f) + 1e-5f);
  float sf = mask[(size_t)m * T_ * T_ + r * T_ + r] ? 1.0f : 0.0f;  // seq[r, b=m]
#pragma unroll
  for (int u = 0; u < 8; ++u) {
    int d = t * 8 + u;
    float x = hs[m * 128 + d];
    out[((size_t)r * 16 + m) * 128 + d] = ((x - mean) * rstd * g2[d] + b2[d]) * sf;
  }
}

// ---------------------------------------------------------------------------
// Kernel 5: boolean outputs as floats.
// ---------------------------------------------------------------------------
__global__ void masks_out(const unsigned char* __restrict__ mask, float* __restrict__ out) {
  int idx = blockIdx.x * blockDim.x + threadIdx.x;
  const int o0 = T_ * B_ * D_;               // 262144
  if (idx < 2048) {                          // padding_mask (B,T) = !seq^T
    int b = idx >> 7, t = idx & 127;
    bool seq = mask[(size_t)b * T_ * T_ + t * T_ + t] != 0;
    out[o0 + idx] = seq ? 0.0f : 1.0f;
  } else if (idx < 4096) {                   // sequence_mask (T,B,1) = seq
    int e = idx - 2048;
    int t = e >> 4, b = e & 15;
    bool seq = mask[(size_t)b * T_ * T_ + t * T_ + t] != 0;
    out[o0 + 2048 + e] = seq ? 1.0f : 0.0f;
  } else if (idx < 4224) {                   // global_mask (T,) ones
    out[o0 + 4096 + (idx - 4096)] = 1.0f;
  }
}

extern "C" void kernel_launch(void* const* d_in, const int* in_sizes, int n_in,
                              void* d_out, int out_size, void* d_ws, size_t ws_size,
                              hipStream_t stream) {
  (void)in_sizes; (void)n_in; (void)out_size; (void)ws_size;
  const float* vectors = (const float*)d_in[0];
  const unsigned char* mask = (const unsigned char*)d_in[1];
  const float* W_embed = (const float*)d_in[2];
  const float* b_embed = (const float*)d_in[3];
  const float* ln1_g = (const float*)d_in[4];
  const float* ln1_b = (const float*)d_in[5];
  const float* Wq = (const float*)d_in[6];
  const float* Wk = (const float*)d_in[7];
  const float* Wv = (const float*)d_in[8];
  const float* W_out = (const float*)d_in[9];
  const float* b_out = (const float*)d_in[10];
  const float* ln2_g = (const float*)d_in[11];
  const float* ln2_b = (const float*)d_in[12];

  char* ws = (char*)d_ws;
  const size_t XB = (size_t)T_ * T_ * B_ * D_ * 2;      // 64 MB bf16
  __bf16* x_bf  = (__bf16*)(ws);
  __bf16* K_bf  = (__bf16*)(ws + XB);
  __bf16* V_bf  = (__bf16*)(ws + 2 * XB);
  float*  Q_f   = (float*)(ws + 3 * XB);
  float*  emb_f = (float*)(ws + 3 * XB + (size_t)T_ * B_ * D_ * 4);
  __bf16* Wt    = (__bf16*)(ws + 3 * XB + (size_t)2 * T_ * B_ * D_ * 4);
  __bf16* WqT = Wt;
  __bf16* WkT = Wt + 16384;
  __bf16* WvT = Wt + 32768;
  __bf16* WoT = Wt + 49152;
  float* out = (float*)d_out;

  prep_weights<<<256, 256, 0, stream>>>(Wq, Wk, Wv, W_out, Wt);
  embed_ln1<<<32768, 256, 0, stream>>>(vectors, mask, W_embed, b_embed, ln1_g, ln1_b, x_bf);
  kv_gemm<<<1024, 256, 0, stream>>>(x_bf, WkT, WvT, K_bf, V_bf);
  q_gemm<<<16, 256, 0, stream>>>(x_bf, WqT, Q_f);
  attention<<<128, 256, 0, stream>>>(Q_f, K_bf, V_bf, mask, emb_f);
  outproj_ln2<<<128, 256, 0, stream>>>(emb_f, WoT, b_out, ln2_g, ln2_b, mask, out);
  masks_out<<<17, 256, 0, stream>>>(mask, out);
}
